// BayesianTransformer_56160992363011
// MI455X (gfx1250) — matrix operand order repaired and checked by reading, not compiled, not run on hardware
//
#include <hip/hip_runtime.h>
#include <hip/hip_bf16.h>
#include <cstdint>
#include <cstddef>

// Problem constants (match reference: V,D,S,B / H = 4D)
#define BT_B 4
#define BT_S 2048
#define BT_D 1024
#define BT_H 4096
#define BT_V 32000

typedef __attribute__((ext_vector_type(16))) __bf16 v16bf;
typedef __attribute__((ext_vector_type(8)))  __bf16 v8bf;
typedef __attribute__((ext_vector_type(8)))  float  v8f;

#define FLAG_RELU   1
#define FLAG_CAUSAL 2
#define FLAG_TRANSC 4

// Padded LDS row stride (bf16 elems) for a K=32 tile row; 80B keeps 16B
// alignment for v8bf accesses and spreads banks.
#define LDST 40

// CDNA5 async global->LDS copy (ASYNCcnt-tracked), double buffered.
#ifndef USE_ASYNC_COPY
#define USE_ASYNC_COPY 1
#endif

#if __has_builtin(__builtin_amdgcn_s_wait_asynccnt)
#define WAIT_ASYNC(n) __builtin_amdgcn_s_wait_asynccnt(n)
#else
#define WAIT_ASYNC(n) asm volatile("s_wait_asynccnt %0" ::"n"(n) : "memory")
#endif

__device__ __forceinline__ void async_b128_to_lds(unsigned lds_off,
                                                  const void* gaddr) {
  // GLOBAL_LOAD_ASYNC_TO_LDS_B128 (GV mode): 16B per lane, VDST = LDS byte
  // address, VADDR = 64-bit global address. Tracked with ASYNCcnt.
  asm volatile("global_load_async_to_lds_b128 %0, %1, off"
               :
               : "v"(lds_off), "v"(gaddr)
               : "memory");
}

// ---------------------------------------------------------------------------
// WMMA fragment loads per CDNA5 ISA 7.12.2 (wave32)
// A (16x32 bf16): lanes 0-15 -> M=lane,    K = {0..7, 16..23}
//                 lanes16-31 -> M=lane-16, K = {8..15, 24..31}
__device__ __forceinline__ v16bf frag_a(const __bf16* base, int lane) {
  const int row = lane & 15;
  const int kg  = (lane >> 4) * 8;
  const __bf16* p = base + row * LDST + kg;
  v8bf lo = *(const v8bf*)(p);
  v8bf hi = *(const v8bf*)(p + 16);
  v16bf r;
#pragma unroll
  for (int i = 0; i < 8; ++i) { r[i] = lo[i]; r[i + 8] = hi[i]; }
  return r;
}

// B (32x16 bf16): lanes 0-15 -> N=lane, K={0..7,16..23} ; lanes16-31 -> N=lane-16,
// K={8..15,24..31}. (Bt tile staged [n][k] row-major.)
__device__ __forceinline__ v16bf frag_b(const __bf16* base, int lane) {
  const int row = lane & 15;
  const int kg  = (lane >> 4) * 8;
  const __bf16* p = base + row * LDST + kg;
  v8bf lo = *(const v8bf*)(p);
  v8bf hi = *(const v8bf*)(p + 16);
  v16bf r;
#pragma unroll
  for (int i = 0; i < 8; ++i) { r[i] = lo[i]; r[i + 8] = hi[i]; }
  return r;
}

// ---------------------------------------------------------------------------
// Batched GEMM: C[M,N] = alpha * A[M,K] x Bt[N,K]^T (+ R) [+ReLU]
// Block tile 128x128, 256 threads = 8 waves (4x2), wave tile 32x64 = 2x4 WMMA.
// LDS double-buffered, filled with global_load_async_to_lds_b128.
// Assumes M%128==0, N%128==0, K%32==0 (true for all uses here).
__global__ __launch_bounds__(256)
void gemm_bf16_wmma(const __bf16* __restrict__ A,  long long strideA,
                    const __bf16* __restrict__ Bt, long long strideB,
                    float* __restrict__ Cf, __bf16* __restrict__ Cb,
                    long long strideC,
                    const float* __restrict__ R, long long strideR,
                    int M, int N, int K, float alpha, int flags)
{
  const int m0 = blockIdx.y * 128;
  const int n0 = blockIdx.x * 128;
  // Causal skip: tile entirely above the diagonal contributes nothing.
  if ((flags & FLAG_CAUSAL) && n0 > m0 + 127) return;

  const int z = blockIdx.z;
  A  += (size_t)z * strideA;
  Bt += (size_t)z * strideB;

  __shared__ __bf16 sA[2][128 * LDST];
  __shared__ __bf16 sB[2][128 * LDST];

  const int tid  = threadIdx.x;
  const int lane = tid & 31;
  const int wave = tid >> 5;
  const int wm   = (wave & 3) * 32;   // wave row offset in block tile
  const int wn   = (wave >> 2) * 64;  // wave col offset in block tile

  v8f acc[2][4];
#pragma unroll
  for (int i = 0; i < 2; ++i)
#pragma unroll
    for (int j = 0; j < 4; ++j)
#pragma unroll
      for (int r = 0; r < 8; ++r) acc[i][j][r] = 0.0f;

  // Tile staging map: 128 rows x 32 cols, 16B (8 bf16) per segment.
  // seg = tid + it*256 -> row = seg>>2 (0..127), col = (seg&3)*8.
  const int row0 = tid >> 2;
  const int col0 = (tid & 3) * 8;

#if USE_ASYNC_COPY
  auto issue_tile = [&](int buf, int k0) {
#pragma unroll
    for (int it = 0; it < 2; ++it) {
      const int row = row0 + it * 64;
      async_b128_to_lds((unsigned)(uintptr_t)&sA[buf][row * LDST + col0],
                        A + (size_t)(m0 + row) * K + k0 + col0);
    }
#pragma unroll
    for (int it = 0; it < 2; ++it) {
      const int row = row0 + it * 64;
      async_b128_to_lds((unsigned)(uintptr_t)&sB[buf][row * LDST + col0],
                        Bt + (size_t)(n0 + row) * K + k0 + col0);
    }
  };
#else
  auto issue_tile = [&](int buf, int k0) {
#pragma unroll
    for (int it = 0; it < 2; ++it) {
      const int row = row0 + it * 64;
      *(v8bf*)&sA[buf][row * LDST + col0] =
          *(const v8bf*)(A + (size_t)(m0 + row) * K + k0 + col0);
      *(v8bf*)&sB[buf][row * LDST + col0] =
          *(const v8bf*)(Bt + (size_t)(n0 + row) * K + k0 + col0);
    }
  };
#endif

  issue_tile(0, 0);
  int buf = 0;
  for (int k0 = 0; k0 < K; k0 += 32) {
    const bool last = (k0 + 32 >= K);
    if (!last) issue_tile(buf ^ 1, k0 + 32);   // prefetch next tile into LDS
#if USE_ASYNC_COPY
    if (!last) WAIT_ASYNC(4);   // current tile landed; next 4 still in flight
    else       WAIT_ASYNC(0);
#endif
    __syncthreads();

    v16bf af[2], bq[4];
#pragma unroll
    for (int i = 0; i < 2; ++i)
      af[i] = frag_a(&sA[buf][(wm + i * 16) * LDST], lane);
#pragma unroll
    for (int j = 0; j < 4; ++j)
      bq[j] = frag_b(&sB[buf][(wn + j * 16) * LDST], lane);

#pragma unroll
    for (int i = 0; i < 2; ++i)
#pragma unroll
      for (int j = 0; j < 4; ++j)
        acc[i][j] = __builtin_amdgcn_wmma_f32_16x16x32_bf16(
            false, af[i], false, bq[j], (short)0, acc[i][j], false, false);

    __syncthreads();
    buf ^= 1;
  }

  float*       Cfp = Cf ? Cf + (size_t)z * strideC : nullptr;
  __bf16*      Cbp = Cb ? Cb + (size_t)z * strideC : nullptr;
  const float* Rp  = R  ? R  + (size_t)z * strideR : nullptr;

  // C layout: VGPR r -> M = r + 8*(lane>=16), N = lane%16
  const int nloc = lane & 15;
  const int mh   = (lane >> 4) * 8;
#pragma unroll
  for (int i = 0; i < 2; ++i) {
#pragma unroll
    for (int j = 0; j < 4; ++j) {
      const int gr0 = m0 + wm + i * 16 + mh;
      const int gc  = n0 + wn + j * 16 + nloc;
#pragma unroll
      for (int r = 0; r < 8; ++r) {
        float v = acc[i][j][r] * alpha;
        const size_t idx = (size_t)(gr0 + r) * N + gc;
        if (Rp) v += Rp[idx];
        if (flags & FLAG_RELU) v = fmaxf(v, 0.0f);
        if (Cfp) Cfp[idx] = v;
        if (Cbp) {
          if (flags & FLAG_TRANSC) Cbp[(size_t)gc * M + gr0 + r] = (__bf16)v;
          else                     Cbp[idx] = (__bf16)v;
        }
      }
    }
  }
}

// ---------------------------------------------------------------------------
// h[b,s,:] = emb[tokens[b,s],:] + pe[s,:]  (f32 + bf16 copies)
__global__ __launch_bounds__(256)
void embed_kernel(const int* __restrict__ tokens, const float* __restrict__ emb,
                  const float* __restrict__ pe, float* __restrict__ h,
                  __bf16* __restrict__ hb, int S, int D)
{
  const int bs  = blockIdx.x;
  const int s   = bs % S;
  const int tok = tokens[bs];
  for (int d = threadIdx.x; d < D; d += blockDim.x) {
    const float v = emb[(size_t)tok * D + d] + pe[(size_t)s * D + d];
    h [(size_t)bs * D + d] = v;
    hb[(size_t)bs * D + d] = (__bf16)v;
  }
}

// ---------------------------------------------------------------------------
// out[n,k] = (bf16) in[k,n]   (weight transpose + convert)
__global__ __launch_bounds__(256)
void transpose_convert_kernel(const float* __restrict__ in,
                              __bf16* __restrict__ out, int Kdim, int Ndim)
{
  __shared__ float tile[32][33];
  const int k0 = blockIdx.y * 32, n0 = blockIdx.x * 32;
  const int tx = threadIdx.x & 31, ty = threadIdx.x >> 5;  // 8 rows/step
  for (int i = ty; i < 32; i += 8) {
    const int k = k0 + i, n = n0 + tx;
    tile[i][tx] = (k < Kdim && n < Ndim) ? in[(size_t)k * Ndim + n] : 0.0f;
  }
  __syncthreads();
  for (int i = ty; i < 32; i += 8) {
    const int n = n0 + i, k = k0 + tx;
    if (n < Ndim && k < Kdim) out[(size_t)n * Kdim + k] = (__bf16)tile[tx][i];
  }
}

// ---------------------------------------------------------------------------
// Causal row softmax: one wave32 per row; masked tail written as 0 (bf16).
__global__ __launch_bounds__(256)
void softmax_causal(const float* __restrict__ scores, __bf16* __restrict__ attn,
                    int S)
{
  const int wave = threadIdx.x >> 5, lane = threadIdx.x & 31;
  const int row  = blockIdx.x * 8 + wave;
  const int b    = blockIdx.y;
  const float* src = scores + ((size_t)b * S + row) * S;
  __bf16*      dst = attn   + ((size_t)b * S + row) * S;
  const int valid = row + 1;

  float mx = -3.402823466e38f;
  for (int t = lane; t < valid; t += 32) mx = fmaxf(mx, src[t]);
#pragma unroll
  for (int o = 16; o > 0; o >>= 1) mx = fmaxf(mx, __shfl_xor(mx, o, 32));

  float sum = 0.0f;
  for (int t = lane; t < valid; t += 32) sum += __expf(src[t] - mx);
#pragma unroll
  for (int o = 16; o > 0; o >>= 1) sum += __shfl_xor(sum, o, 32);
  const float inv = 1.0f / sum;

  for (int t = lane; t < S; t += 32)
    dst[t] = (t < valid) ? (__bf16)(__expf(src[t] - mx) * inv) : (__bf16)0.0f;
}

// ---------------------------------------------------------------------------
// logits[b,v] = h[b,S-1,:] . w_out_t[v,:]   (memory-bound, ~0.26 GFLOP)
__global__ __launch_bounds__(256)
void logits_kernel(const float* __restrict__ h, const __bf16* __restrict__ woutT,
                   float* __restrict__ out, int S, int D, int V)
{
  __shared__ float hv[BT_D];
  const int b = blockIdx.y;
  const float* hrow = h + ((size_t)b * S + (S - 1)) * D;
  for (int d = threadIdx.x; d < D; d += blockDim.x) hv[d] = hrow[d];
  __syncthreads();
  const int v = blockIdx.x * blockDim.x + threadIdx.x;
  if (v >= V) return;
  const __bf16* w = woutT + (size_t)v * D;
  float s = 0.0f;
  for (int d = 0; d < D; d += 8) {
    v8bf wv = *(const v8bf*)(w + d);
#pragma unroll
    for (int i = 0; i < 8; ++i) s += hv[d + i] * (float)wv[i];
  }
  out[(size_t)b * V + v] = s;
}

// ---------------------------------------------------------------------------
extern "C" void kernel_launch(void* const* d_in, const int* in_sizes, int n_in,
                              void* d_out, int out_size, void* d_ws,
                              size_t ws_size, hipStream_t stream)
{
  (void)in_sizes; (void)n_in; (void)out_size; (void)ws_size;

  const int*   tokens = (const int*)d_in[0];
  const float* emb    = (const float*)d_in[1];
  const float* pe     = (const float*)d_in[2];

  // ---- workspace layout (256B aligned slabs) ----
  char* ws = (char*)d_ws;
  size_t off = 0;
  auto take = [&](size_t bytes) -> char* {
    char* p = ws + off;
    off += (bytes + 255) & ~(size_t)255;
    return p;
  };
  const size_t nBSD = (size_t)BT_B * BT_S * BT_D;
  const size_t nBSS = (size_t)BT_B * BT_S * BT_S;

  float*  h      = (float*) take(nBSD * 4);
  __bf16* hb     = (__bf16*)take(nBSD * 2);
  float*  hres   = (float*) take(nBSD * 4);
  __bf16* hresb  = (__bf16*)take(nBSD * 2);
  __bf16* xk     = (__bf16*)take(nBSD * 2);
  __bf16* xvT    = (__bf16*)take(nBSD * 2);   // [D,S] per batch
  __bf16* attnv  = (__bf16*)take(nBSD * 2);
  float*  scores = (float*) take(nBSS * 4);   // reused as t1 (S*H bf16, same bytes)
  __bf16* attn   = (__bf16*)take(nBSS * 2);
  __bf16* t1     = (__bf16*)scores;           // B*S*H*2 == nBSS*4 bytes

  // weights -> bf16, transposed to [N,K]
  const int wKs[11] = {BT_D, BT_D, BT_D, BT_D, BT_H,
                       BT_D, BT_D, BT_D, BT_D, BT_H, BT_D};
  const int wNs[11] = {BT_D, BT_D, BT_D, BT_H, BT_D,
                       BT_D, BT_D, BT_D, BT_H, BT_D, BT_V};
  __bf16* wT[11];
  for (int i = 0; i < 11; ++i)
    wT[i] = (__bf16*)take((size_t)wKs[i] * wNs[i] * 2);
  for (int i = 0; i < 11; ++i) {
    dim3 g((wNs[i] + 31) / 32, (wKs[i] + 31) / 32);
    transpose_convert_kernel<<<g, 256, 0, stream>>>(
        (const float*)d_in[3 + i], wT[i], wKs[i], wNs[i]);
  }

  embed_kernel<<<BT_B * BT_S, 256, 0, stream>>>(tokens, emb, pe, h, hb,
                                                BT_S, BT_D);

  auto gemm = [&](const __bf16* Ap, long long sA, const __bf16* Bp, long long sB,
                  float* Cfp, __bf16* Cbp, long long sC,
                  const float* Rp, long long sR,
                  int M, int N, int K, float alpha, int flags) {
    dim3 grid(N / 128, M / 128, BT_B);
    gemm_bf16_wmma<<<grid, 256, 0, stream>>>(Ap, sA, Bp, sB, Cfp, Cbp, sC,
                                             Rp, sR, M, N, K, alpha, flags);
  };

  const long long nSD = (long long)BT_S * BT_D;
  const long long nSS = (long long)BT_S * BT_S;
  const long long nSH = (long long)BT_S * BT_H;

  for (int l = 0; l < 2; ++l) {
    const __bf16* wk = wT[l * 5 + 0];
    const __bf16* wv = wT[l * 5 + 1];
    const __bf16* wo = wT[l * 5 + 2];
    const __bf16* w1 = wT[l * 5 + 3];
    const __bf16* w2 = wT[l * 5 + 4];

    // xk = h @ Wk ; xv^T (stored [D,S]) = (h @ Wv)^T
    gemm(hb, nSD, wk, 0, nullptr, xk, nSD, nullptr, 0,
         BT_S, BT_D, BT_D, 1.0f, 0);
    gemm(hb, nSD, wv, 0, nullptr, xvT, nSD, nullptr, 0,
         BT_S, BT_D, BT_D, 1.0f, FLAG_TRANSC);
    // scores = (q @ k^T)/sqrt(D), skip fully-masked tiles
    gemm(hb, nSD, xk, nSD, scores, nullptr, nSS, nullptr, 0,
         BT_S, BT_S, BT_D, 0.03125f, FLAG_CAUSAL);
    softmax_causal<<<dim3(BT_S / 8, BT_B), 256, 0, stream>>>(scores, attn, BT_S);
    // attnv = attn @ xv
    gemm(attn, nSS, xvT, nSD, nullptr, attnv, nSD, nullptr, 0,
         BT_S, BT_D, BT_S, 1.0f, 0);
    // h_res = h + attnv @ Wo
    gemm(attnv, nSD, wo, 0, hres, hresb, nSD, h, nSD,
         BT_S, BT_D, BT_D, 1.0f, 0);
    // t1 = relu(h_res @ W1)
    gemm(hresb, nSD, w1, 0, nullptr, t1, nSH, nullptr, 0,
         BT_S, BT_H, BT_D, 1.0f, FLAG_RELU);
    // h = h_res + t1 @ W2
    gemm(t1, nSH, w2, 0, h, hb, nSD, hres, nSD,
         BT_S, BT_D, BT_H, 1.0f, 0);
  }

  logits_kernel<<<dim3((BT_V + 255) / 256, BT_B), 256, 0, stream>>>(
      h, wT[10], (float*)d_out, BT_S, BT_D, BT_V);
}
